// Decoder_3289944948995
// MI455X (gfx1250) — matrix operand order repaired and checked by reading, not compiled, not run on hardware
//
#include <hip/hip_runtime.h>
#include <hip/hip_bf16.h>

typedef float          v2f   __attribute__((ext_vector_type(2)));
typedef float          v8f   __attribute__((ext_vector_type(8)));
typedef __bf16         v16bf __attribute__((ext_vector_type(16)));
typedef unsigned short v8us  __attribute__((ext_vector_type(8)));
typedef unsigned short v16us __attribute__((ext_vector_type(16)));

#define HIDDEN  512
#define TWOH    1024
#define NCHUNK  128                 // layer-1 columns per pass
#define NTILES  (NCHUNK / 16)       // 8 WMMA tiles per pass
#define NPASS   (HIDDEN / NCHUNK)   // 4 passes

// ---------------------------------------------------------------------------
// Split-precision prepass: x -> (hi = bf16(x), lo = bf16(x - hi))
// ---------------------------------------------------------------------------
__global__ void split_f32_bf16(const float* __restrict__ src,
                               unsigned short* __restrict__ hi,
                               unsigned short* __restrict__ lo, int n)
{
    int i = blockIdx.x * blockDim.x + threadIdx.x;
    if (i < n) {
        float  x = src[i];
        __bf16 h = (__bf16)x;
        __bf16 l = (__bf16)(x - (float)h);
        hi[i] = __builtin_bit_cast(unsigned short, h);
        lo[i] = __builtin_bit_cast(unsigned short, l);
    }
}

// Build a v16bf A-fragment: per-lane K = {k..k+7} and {k+16..k+23} (+8 for hi half-wave)
__device__ inline v16bf ldA(const unsigned short* p)
{
    v8us c0 = *(const v8us*)(p);
    v8us c1 = *(const v8us*)(p + 16);
    v16us w = __builtin_shufflevector(c0, c1, 0,1,2,3,4,5,6,7,8,9,10,11,12,13,14,15);
    return __builtin_bit_cast(v16bf, w);
}
// Build a v16bf B-fragment: per-lane K = {k..k+7} and {k+16..k+23} (+8 for hi half-wave)
__device__ inline v16bf ldB(const unsigned short* p)
{
    v8us c0 = *(const v8us*)(p);
    v8us c1 = *(const v8us*)(p + 16);
    v16us w = __builtin_shufflevector(c0, c1, 0,1,2,3,4,5,6,7,8,9,10,11,12,13,14,15);
    return __builtin_bit_cast(v16bf, w);
}

// ---------------------------------------------------------------------------
// bf16x3 fused kernel: one wave = 16 edges end-to-end.
// D = Ahi*Bhi + Ahi*Blo + Alo*Bhi  (fp32 accumulate) ~ fp32-quality GEMM
// ---------------------------------------------------------------------------
__global__ __launch_bounds__(128, 1)
void fused_edge_decoder_bf16x3(const unsigned short* __restrict__ zdh,
                               const unsigned short* __restrict__ zdl,
                               const unsigned short* __restrict__ zrh,
                               const unsigned short* __restrict__ zrl,
                               const unsigned short* __restrict__ w1h,
                               const unsigned short* __restrict__ w1l,
                               const int*   __restrict__ row,
                               const int*   __restrict__ col,
                               const float* __restrict__ b1,
                               const float* __restrict__ W2,
                               const float* __restrict__ b2p,
                               float* __restrict__ out, int E)
{
    const int lane = threadIdx.x & 31;
    const int wave = blockIdx.x * (blockDim.x >> 5) + (threadIdx.x >> 5);
    const int e0   = wave * 16;

    const int ml = lane & 15;          // A: edge row (M); B: column (N)
    const int ak = (lane >> 4) * 8;    // A K sub-offset per half-wave
    const int bk = (lane >> 4) * 8;    // B K sub-offset per half-wave

    int e  = e0 + ml;
    int ec = (e < E) ? e : (E - 1);
    const unsigned short* aDh = zdh + (size_t)row[ec] * HIDDEN + ak;
    const unsigned short* aDl = zdl + (size_t)row[ec] * HIDDEN + ak;
    const unsigned short* aRh = zrh + (size_t)col[ec] * HIDDEN + ak;
    const unsigned short* aRl = zrl + (size_t)col[ec] * HIDDEN + ak;

    float acc2[8];
    #pragma unroll
    for (int v = 0; v < 8; ++v) acc2[v] = 0.0f;

    #pragma unroll 1
    for (int pass = 0; pass < NPASS; ++pass) {
        const int n0 = pass * NCHUNK;
        const unsigned short* bH = w1h + (size_t)(n0 + ml) * TWOH + bk;
        const unsigned short* bL = w1l + (size_t)(n0 + ml) * TWOH + bk;

        v8f acc[NTILES];
        #pragma unroll
        for (int t = 0; t < NTILES; ++t) acc[t] = (v8f){};

        // ---- K = 0..511 : drug half of the concatenated feature ----
        #pragma unroll 2
        for (int k = 0; k < HIDDEN; k += 32) {
            v16bf ah = ldA(aDh + k);
            v16bf al = ldA(aDl + k);
            #pragma unroll
            for (int t = 0; t < NTILES; ++t) {
                const size_t off = (size_t)t * 16 * TWOH + k;
                v16bf bh = ldB(bH + off);
                v16bf bl = ldB(bL + off);
                acc[t] = __builtin_amdgcn_wmma_f32_16x16x32_bf16(
                    false, ah, false, bh, (short)0, acc[t], false, false);
                acc[t] = __builtin_amdgcn_wmma_f32_16x16x32_bf16(
                    false, ah, false, bl, (short)0, acc[t], false, false);
                acc[t] = __builtin_amdgcn_wmma_f32_16x16x32_bf16(
                    false, al, false, bh, (short)0, acc[t], false, false);
            }
        }
        // ---- K = 512..1023 : reaction half ----
        #pragma unroll 2
        for (int k = 0; k < HIDDEN; k += 32) {
            v16bf ah = ldA(aRh + k);
            v16bf al = ldA(aRl + k);
            #pragma unroll
            for (int t = 0; t < NTILES; ++t) {
                const size_t off = (size_t)t * 16 * TWOH + HIDDEN + k;
                v16bf bh = ldB(bH + off);
                v16bf bl = ldB(bL + off);
                acc[t] = __builtin_amdgcn_wmma_f32_16x16x32_bf16(
                    false, ah, false, bh, (short)0, acc[t], false, false);
                acc[t] = __builtin_amdgcn_wmma_f32_16x16x32_bf16(
                    false, ah, false, bl, (short)0, acc[t], false, false);
                acc[t] = __builtin_amdgcn_wmma_f32_16x16x32_bf16(
                    false, al, false, bh, (short)0, acc[t], false, false);
            }
        }

        // ---- fused layer-2 epilogue (fp32): acc2[m] += relu(h+b1[n]) * W2[n] ----
        #pragma unroll
        for (int t = 0; t < NTILES; ++t) {
            const int n    = n0 + t * 16 + ml;
            const float b1v = b1[n];
            const float w2v = W2[n];
            #pragma unroll
            for (int v = 0; v < 8; ++v) {
                float h = acc[t][v] + b1v;
                h = fmaxf(h, 0.0f);
                acc2[v] = __builtin_fmaf(h, w2v, acc2[v]);
            }
        }
    }

    #pragma unroll
    for (int off = 8; off > 0; off >>= 1) {
        #pragma unroll
        for (int v = 0; v < 8; ++v)
            acc2[v] += __shfl_xor(acc2[v], off, 16);
    }

    if (ml == 0) {
        const float b2v = b2p[0];
        const int base = e0 + (lane >> 4) * 8;
        #pragma unroll
        for (int j = 0; j < 8; ++j) {
            int ee = base + j;
            if (ee < E) out[ee] = acc2[j] + b2v;
        }
    }
}

// ---------------------------------------------------------------------------
// fp32 WMMA fallback (used only if ws_size is too small for the bf16 splits)
// ---------------------------------------------------------------------------
__global__ __launch_bounds__(128, 1)
void fused_edge_decoder_f32(const float* __restrict__ zd,
                            const float* __restrict__ zr,
                            const int*   __restrict__ row,
                            const int*   __restrict__ col,
                            const float* __restrict__ W1,
                            const float* __restrict__ b1,
                            const float* __restrict__ W2,
                            const float* __restrict__ b2p,
                            float* __restrict__ out, int E)
{
    const int lane = threadIdx.x & 31;
    const int wave = blockIdx.x * (blockDim.x >> 5) + (threadIdx.x >> 5);
    const int e0   = wave * 16;
    const int ml = lane & 15;
    const int kh = (lane >> 4) * 2;

    int e  = e0 + ml;
    int ec = (e < E) ? e : (E - 1);
    const float* aDrug = zd + (size_t)row[ec] * HIDDEN + kh;
    const float* aReac = zr + (size_t)col[ec] * HIDDEN + kh;

    float acc2[8];
    #pragma unroll
    for (int v = 0; v < 8; ++v) acc2[v] = 0.0f;

    #pragma unroll 1
    for (int pass = 0; pass < NPASS; ++pass) {
        const int n0 = pass * NCHUNK;
        const float* bBase = W1 + (size_t)(n0 + ml) * TWOH + kh;

        v8f acc[NTILES];
        #pragma unroll
        for (int t = 0; t < NTILES; ++t) acc[t] = (v8f){};

        #pragma unroll 4
        for (int k = 0; k < HIDDEN; k += 4) {
            v2f a = *(const v2f*)(aDrug + k);
            #pragma unroll
            for (int t = 0; t < NTILES; ++t) {
                v2f b = *(const v2f*)(bBase + (size_t)t * 16 * TWOH + k);
                acc[t] = __builtin_amdgcn_wmma_f32_16x16x4_f32(
                    false, a, false, b, (short)0, acc[t], false, false);
            }
        }
        #pragma unroll 4
        for (int k = 0; k < HIDDEN; k += 4) {
            v2f a = *(const v2f*)(aReac + k);
            #pragma unroll
            for (int t = 0; t < NTILES; ++t) {
                v2f b = *(const v2f*)(bBase + (size_t)t * 16 * TWOH + HIDDEN + k);
                acc[t] = __builtin_amdgcn_wmma_f32_16x16x4_f32(
                    false, a, false, b, (short)0, acc[t], false, false);
            }
        }

        #pragma unroll
        for (int t = 0; t < NTILES; ++t) {
            const int n    = n0 + t * 16 + ml;
            const float b1v = b1[n];
            const float w2v = W2[n];
            #pragma unroll
            for (int v = 0; v < 8; ++v) {
                float h = acc[t][v] + b1v;
                h = fmaxf(h, 0.0f);
                acc2[v] = __builtin_fmaf(h, w2v, acc2[v]);
            }
        }
    }

    #pragma unroll
    for (int off = 8; off > 0; off >>= 1) {
        #pragma unroll
        for (int v = 0; v < 8; ++v)
            acc2[v] += __shfl_xor(acc2[v], off, 16);
    }

    if (ml == 0) {
        const float b2v = b2p[0];
        const int base = e0 + (lane >> 4) * 8;
        #pragma unroll
        for (int j = 0; j < 8; ++j) {
            int ee = base + j;
            if (ee < E) out[ee] = acc2[j] + b2v;
        }
    }
}

// ---------------------------------------------------------------------------
extern "C" void kernel_launch(void* const* d_in, const int* in_sizes, int n_in,
                              void* d_out, int out_size, void* d_ws, size_t ws_size,
                              hipStream_t stream) {
    const float* zd  = (const float*)d_in[0];
    const float* zr  = (const float*)d_in[1];
    const int*   row = (const int*)  d_in[2];
    const int*   col = (const int*)  d_in[3];
    const float* W1  = (const float*)d_in[4];
    const float* b1  = (const float*)d_in[5];
    const float* W2  = (const float*)d_in[6];
    const float* b2  = (const float*)d_in[7];
    float* out = (float*)d_out;

    const int    E   = in_sizes[2];
    const size_t nzd = (size_t)in_sizes[0];
    const size_t nzr = (size_t)in_sizes[1];
    const size_t nW1 = (size_t)in_sizes[4];

    const int waves  = (E + 15) / 16;
    const int blocks = (waves + 3) / 4;     // 4 waves (128 threads) per block

    // Workspace layout (bf16 element offsets, 256B-aligned)
    auto aup = [](size_t x) { return (x + 127) & ~(size_t)127; };
    size_t oW1h = 0;
    size_t oW1l = aup(oW1h + nW1);
    size_t ozdh = aup(oW1l + nW1);
    size_t ozdl = aup(ozdh + nzd);
    size_t ozrh = aup(ozdl + nzd);
    size_t ozrl = aup(ozrh + nzr);
    size_t needBytes = aup(ozrl + nzr) * sizeof(unsigned short);

    if (ws_size >= needBytes) {
        unsigned short* ws = (unsigned short*)d_ws;
        split_f32_bf16<<<(int)((nW1 + 255) / 256), 256, 0, stream>>>(W1, ws + oW1h, ws + oW1l, (int)nW1);
        split_f32_bf16<<<(int)((nzd + 255) / 256), 256, 0, stream>>>(zd, ws + ozdh, ws + ozdl, (int)nzd);
        split_f32_bf16<<<(int)((nzr + 255) / 256), 256, 0, stream>>>(zr, ws + ozrh, ws + ozrl, (int)nzr);

        fused_edge_decoder_bf16x3<<<blocks, 128, 0, stream>>>(
            ws + ozdh, ws + ozdl, ws + ozrh, ws + ozrl, ws + oW1h, ws + oW1l,
            row, col, b1, W2, b2, out, E);
    } else {
        fused_edge_decoder_f32<<<blocks, 128, 0, stream>>>(
            zd, zr, row, col, W1, b1, W2, b2, out, E);
    }
}
